// DRNN_88751204204780
// MI455X (gfx1250) — matrix operand order repaired and checked by reading, not compiled, not run on hardware
//
#include <hip/hip_runtime.h>
#include <hip/hip_bf16.h>
#include <stdint.h>
#include <stddef.h>

// ---------------------------------------------------------------------------
// Dilated RNN (B=256, S=2048, IN=OUT=64, H=512, dil 1/2/7) for gfx1250.
// Persistent workgroup per 16-row batch tile; bf16 WMMA (16x16x32) with f32
// accumulate; recurrent state in LDS; bf16 weights in d_ws (L2 resident);
// double-buffered x tile hides HBM latency behind each step's compute.
// ---------------------------------------------------------------------------

typedef __attribute__((ext_vector_type(16))) __bf16   v16bf;
typedef __attribute__((ext_vector_type(8)))  float    v8f;
typedef __attribute__((ext_vector_type(4)))  uint32_t uint4v;

#define WAVES   16
#define BLOCK   (WAVES * 32)
#define Hdim    512
#define INdim   64
#define OUTdim  64
#define SEQ     2048
#define BATCH   256

// bf16-weight offsets inside d_ws (in bf16 elements)
#define OFF_WI1 0u
#define OFF_WH1 32768u
#define OFF_WI2 294912u
#define OFF_WH2 557056u
#define OFF_WI3 819200u
#define OFF_WH3 1081344u
#define OFF_WO1 1343488u
#define OFF_WO2 1376256u
#define OFF_WO3 1409024u
#define WTS_ELEMS 1441792u

// LDS layout (bytes)
#define LDS_XT    0u        // bf16 [2][16][64]  (double buffer)  -> 4096
#define LDS_H1    4096u     // bf16 [2][16][512]                  -> 36864
#define LDS_H2    36864u    // bf16 [3][16][512]                  -> 86016
#define LDS_H3    86016u    // bf16 [8][16][512]                  -> 217088
#define LDS_SBUF  217088u   // f32  [8][16][64]                   -> 249856
#define LDS_O1    249856u   // f32  [16][64]                      -> 253952
#define LDS_O2    253952u   // f32  [16][64]                      -> 258048
#define LDS_O3    258048u   // f32  [16][64]                      -> 262144
#define LDS_TOTAL 262144u

// ---------------- fragment loaders (per ISA 16-bit WMMA layouts) -----------

// A matrix 16x32: lane holds row (lane&15); elems 0..7 = K k0+kh*8..,
// elems 8..15 = K k0+16+kh*8.. (kh = lane>>4). Two 16B LDS loads.
__device__ __forceinline__ v16bf loadA(const __bf16* base, int ldk, int k0, int lane) {
    const int row = lane & 15;
    const int kh8 = (lane >> 4) << 3;
    union { v16bf v; uint4v q[2]; } f;
    const __bf16* p = base + row * ldk + k0 + kh8;
    f.q[0] = *(const uint4v*)(p);
    f.q[1] = *(const uint4v*)(p + 16);
    return f.v;
}

// B matrix 32x16: lane holds column (lane&15); elems 0..7 = K k0+kh*8..,
// elems 8..15 = K k0+16+kh*8.. (kh = lane>>4). W row-major [N][K] -> two 16B loads.
__device__ __forceinline__ v16bf loadB(const __bf16* W, int ldk, int n0, int k0, int lane) {
    const int n    = lane & 15;
    const int kh16 = (lane >> 4) << 3;
    union { v16bf v; uint4v q[2]; } f;
    const __bf16* p = W + (size_t)(n0 + n) * ldk + k0 + kh16;
    f.q[0] = *(const uint4v*)(p);
    f.q[1] = *(const uint4v*)(p + 16);
    return f.v;
}

__device__ __forceinline__ v8f wmma_bf16(v16bf a, v16bf b, v8f c) {
    return __builtin_amdgcn_wmma_f32_16x16x32_bf16(false, a, false, b, (short)0, c,
                                                   false, false);
}

// One A-fragment feeds two N-tiles (back-to-back WMMA for ILP).
__device__ __forceinline__ void mm_pass2(const __bf16* A, int K, const __bf16* W,
                                         v8f& acc0, v8f& acc1, int n0, int lane) {
#pragma unroll 4
    for (int k0 = 0; k0 < K; k0 += 32) {
        v16bf a  = loadA(A, K, k0, lane);
        v16bf b0 = loadB(W, K, n0,      k0, lane);
        v16bf b1 = loadB(W, K, n0 + 16, k0, lane);
        acc0 = wmma_bf16(a, b0, acc0);
        acc1 = wmma_bf16(a, b1, acc1);
    }
}

__device__ __forceinline__ void mm_pass1(const __bf16* A, int K, const __bf16* W,
                                         v8f& acc, int n0, int lane) {
#pragma unroll 4
    for (int k0 = 0; k0 < K; k0 += 32) {
        v16bf a = loadA(A, K, k0, lane);
        v16bf b = loadB(W, K, n0, k0, lane);
        acc = wmma_bf16(a, b, acc);
    }
}

// One recurrent layer: Hout = tanh(Ain@Wi^T + Hprev@Wh^T + (bi+bh))
// Each wave produces columns [wave*32, wave*32+32). Biases pre-hoisted.
__device__ __forceinline__ void layer(const __bf16* Ain, int Kin, const __bf16* Wi,
                                      const __bf16* Hprev, const __bf16* Wh,
                                      float bA, float bB,
                                      __bf16* Hout, int wave, int lane) {
    const int n  = lane & 15;
    const int kh = lane >> 4;
    const int n0 = wave << 5;  // 2 tiles of 16
    v8f acc0 = {bA, bA, bA, bA, bA, bA, bA, bA};
    v8f acc1 = {bB, bB, bB, bB, bB, bB, bB, bB};
    mm_pass2(Ain,   Kin,  Wi, acc0, acc1, n0, lane);
    mm_pass2(Hprev, Hdim, Wh, acc0, acc1, n0, lane);
#pragma unroll
    for (int g = 0; g < 8; ++g) {
        const int m = g + (kh << 3);  // C/D layout: vgpr g, rows split by lane>>4
        Hout[m * Hdim + n0 + n]      = (__bf16)tanhf(acc0[g]);
        Hout[m * Hdim + n0 + 16 + n] = (__bf16)tanhf(acc1[g]);
    }
}

// --------------------------------- prep ------------------------------------

__global__ void cvt_f32_bf16(const float* __restrict__ src, __bf16* __restrict__ dst,
                             int n) {
    int i = blockIdx.x * blockDim.x + threadIdx.x;
    if (i < n) dst[i] = (__bf16)src[i];
}

__global__ void zero_losses(float* p) {
    if (threadIdx.x < 3) p[threadIdx.x] = 0.0f;
}

// ------------------------------ main kernel ---------------------------------

__global__ __launch_bounds__(BLOCK, 1)
void drnn_persistent(const float* __restrict__ x,
                     const float* __restrict__ bi1, const float* __restrict__ bh1,
                     const float* __restrict__ bi2, const float* __restrict__ bh2,
                     const float* __restrict__ bi3, const float* __restrict__ bh3,
                     const float* __restrict__ bo1, const float* __restrict__ bo2,
                     const float* __restrict__ bo3,
                     const __bf16* __restrict__ wts,
                     float* __restrict__ out, float* __restrict__ losses) {
    extern __shared__ char smem[];
    __bf16* xt   = (__bf16*)(smem + LDS_XT);   // 2 buffers of 16x64
    __bf16* h1r  = (__bf16*)(smem + LDS_H1);
    __bf16* h2r  = (__bf16*)(smem + LDS_H2);
    __bf16* h3r  = (__bf16*)(smem + LDS_H3);
    float*  sbuf = (float*)(smem + LDS_SBUF);
    float*  o1b  = (float*)(smem + LDS_O1);
    float*  o2b  = (float*)(smem + LDS_O2);
    float*  o3b  = (float*)(smem + LDS_O3);

    const __bf16* Wi1 = wts + OFF_WI1;
    const __bf16* Wh1 = wts + OFF_WH1;
    const __bf16* Wi2 = wts + OFF_WI2;
    const __bf16* Wh2 = wts + OFF_WH2;
    const __bf16* Wi3 = wts + OFF_WI3;
    const __bf16* Wh3 = wts + OFF_WH3;
    const __bf16* Wo1 = wts + OFF_WO1;
    const __bf16* Wo2 = wts + OFF_WO2;
    const __bf16* Wo3 = wts + OFF_WO3;

    const int tid  = threadIdx.x;
    const int lane = tid & 31;
    const int wave = tid >> 5;
    const int b0   = blockIdx.x * 16;  // 16 batch rows per workgroup

    // ---- loop-invariant hoisting -----------------------------------------
    const int n  = lane & 15;
    const int cA = (wave << 5) + n;      // this lane's column in tile 0
    const int cB = cA + 16;              // and in tile 1
    const float b1A = bi1[cA] + bh1[cA], b1B = bi1[cB] + bh1[cB];
    const float b2A = bi2[cA] + bh2[cA], b2B = bi2[cB] + bh2[cB];
    const float b3A = bi3[cA] + bh3[cA], b3B = bi3[cB] + bh3[cB];

    // phase-D (output head) selection, constant per wave
    const int whichD = wave >> 2;                // 0:o1 1:o2 2:o3 (3: idle)
    const int n0D    = (wave & 3) << 4;
    const __bf16* WselD = (whichD == 0) ? Wo1 : (whichD == 1) ? Wo2 : Wo3;
    float*        obufD = (whichD == 0) ? o1b : (whichD == 1) ? o2b : o3b;
    const float*  bselD = (whichD == 0) ? bo1 : (whichD == 1) ? bo2 : bo3;
    const float   biasD = bselD[n0D + n];

    // per-thread x / out element mapping (2 elems of the 16x64 tile each)
    const int e0 = tid, e1 = tid + BLOCK;
    const int r0 = e0 >> 6, c0 = e0 & 63;
    const int r1 = e1 >> 6, c1 = e1 & 63;
    const float* xp0 = x + (size_t)(b0 + r0) * SEQ * INdim + c0;
    const float* xp1 = x + (size_t)(b0 + r1) * SEQ * INdim + c1;
    float* op0 = out + (size_t)(b0 + r0) * SEQ * OUTdim + c0;
    float* op1 = out + (size_t)(b0 + r1) * SEQ * OUTdim + c1;

    // zero h-rings + seasonal cache (initial hidden states / cache are zero)
    {
        uint32_t* z = (uint32_t*)(smem + LDS_H1);
        const int nz = (LDS_O1 - LDS_H1) / 4;
        for (int i = tid; i < nz; i += BLOCK) z[i] = 0u;
    }
    // stage x(0) into xt buffer 0
    {
        const float v0 = __builtin_nontemporal_load(xp0);
        const float v1 = __builtin_nontemporal_load(xp1);
        xt[e0] = (__bf16)v0;
        xt[e1] = (__bf16)v1;
    }
    __syncthreads();

    float rl_acc = 0.0f, sl_acc = 0.0f, ml_acc = 0.0f;

    for (int t = 0; t < SEQ; ++t) {
        // ---- issue NT prefetch of x(t+1); consumed in the epilogue ----
        float nx0 = 0.0f, nx1 = 0.0f;
        if (t + 1 < SEQ) {
            nx0 = __builtin_nontemporal_load(xp0 + (size_t)(t + 1) * INdim);
            nx1 = __builtin_nontemporal_load(xp1 + (size_t)(t + 1) * INdim);
        }

        const __bf16* xcur = xt + (size_t)(t & 1) * 16 * INdim;

        // ring slots: write t%(d+1), read (t-d)%(d+1) == (t+1)%(d+1)
        const __bf16* h1_prev = h1r + (size_t)((t + 1) & 1) * 16 * Hdim;
        __bf16*       h1_cur  = h1r + (size_t)(t & 1)       * 16 * Hdim;
        const __bf16* h2_prev = h2r + (size_t)((t + 1) % 3) * 16 * Hdim;
        __bf16*       h2_cur  = h2r + (size_t)(t % 3)       * 16 * Hdim;
        const __bf16* h3_prev = h3r + (size_t)((t + 1) & 7) * 16 * Hdim;
        __bf16*       h3_cur  = h3r + (size_t)(t & 7)       * 16 * Hdim;

        // ---- layer 1..3 (barrier-separated: each consumes full prev h) ----
        layer(xcur,   INdim, Wi1, h1_prev, Wh1, b1A, b1B, h1_cur, wave, lane);
        __syncthreads();
        layer(h1_cur, Hdim,  Wi2, h2_prev, Wh2, b2A, b2B, h2_cur, wave, lane);
        __syncthreads();
        layer(h2_cur, Hdim,  Wi3, h3_prev, Wh3, b3A, b3B, h3_cur, wave, lane);
        __syncthreads();

        // ---- output heads: 12 waves, one 16-col tile each ----
        if (wave < 12) {
            const __bf16* Hsrc = (whichD == 0) ? h1_cur
                               : (whichD == 1) ? h2_cur : h3_cur;
            const int kh = lane >> 4;
            v8f acc = {biasD, biasD, biasD, biasD, biasD, biasD, biasD, biasD};
            mm_pass1(Hsrc, Hdim, WselD, acc, n0D, lane);
#pragma unroll
            for (int g = 0; g < 8; ++g)
                obufD[(g + (kh << 3)) * OUTdim + n0D + n] = tanhf(acc[g]);
        }
        __syncthreads();

        // ---- epilogue: out = o1+o2+o3, losses, caches ----
        const int slotw = t & 7;         // seasonal write slot
        const int slotr = (t + 1) & 7;   // o2(t-7) slot
        {
            const float v1a = o1b[e0], v2a = o2b[e0], v3a = o3b[e0];
            const float v1b = o1b[e1], v2b = o2b[e1], v3b = o3b[e1];
            __builtin_nontemporal_store(v1a + v2a + v3a, op0 + (size_t)t * OUTdim);
            __builtin_nontemporal_store(v1b + v2b + v3b, op1 + (size_t)t * OUTdim);
            rl_acc += v1a * v1a + v1b * v1b;
            if (t >= 7) {
                const float da = v2a - sbuf[slotr * 1024 + e0];
                const float db = v2b - sbuf[slotr * 1024 + e1];
                sl_acc += da * da + db * db;
            }
            sbuf[slotw * 1024 + e0] = v2a;
            sbuf[slotw * 1024 + e1] = v2b;
        }
        if (tid < 16) {  // unbiased per-row variance of o3 over OUT=64
            float s1 = 0.0f, s2 = 0.0f;
#pragma unroll 8
            for (int c = 0; c < 64; ++c) {
                const float v = o3b[tid * 64 + c];
                s1 += v; s2 += v * v;
            }
            ml_acc += (s2 - s1 * s1 * (1.0f / 64.0f)) * (1.0f / 63.0f);
        }
        // commit prefetched x(t+1) into the alternate buffer
        if (t + 1 < SEQ) {
            __bf16* xnext = xt + (size_t)((t + 1) & 1) * 16 * INdim;
            xnext[e0] = (__bf16)nx0;
            xnext[e1] = (__bf16)nx1;
        }
        __syncthreads();
    }

    // ---- loss reduction: wave shuffle -> LDS -> one atomicAdd each ----
#pragma unroll
    for (int off = 16; off > 0; off >>= 1) {
        rl_acc += __shfl_down(rl_acc, off);
        sl_acc += __shfl_down(sl_acc, off);
        ml_acc += __shfl_down(ml_acc, off);
    }
    float* red = o1b;  // reuse
    if (lane == 0) {
        red[wave] = rl_acc; red[64 + wave] = sl_acc; red[128 + wave] = ml_acc;
    }
    __syncthreads();
    if (tid == 0) {
        float rl = 0.0f, sl = 0.0f, ml = 0.0f;
        for (int w = 0; w < WAVES; ++w) {
            rl += red[w]; sl += red[64 + w]; ml += red[128 + w];
        }
        atomicAdd(&losses[0], rl * (1.0f / (float)(BATCH * OUTdim)));
        atomicAdd(&losses[1], sl * (1.0f / (float)(BATCH * OUTdim)));
        atomicAdd(&losses[2], ml * (1.0f / (float)BATCH));
    }
}

// ------------------------------- launcher -----------------------------------

extern "C" void kernel_launch(void* const* d_in, const int* in_sizes, int n_in,
                              void* d_out, int out_size, void* d_ws, size_t ws_size,
                              hipStream_t stream) {
    (void)n_in; (void)in_sizes;
    if (ws_size < (size_t)WTS_ELEMS * sizeof(__bf16)) return;

    const float* x   = (const float*)d_in[0];
    const float* bi1 = (const float*)d_in[2];
    const float* bh1 = (const float*)d_in[4];
    const float* bi2 = (const float*)d_in[6];
    const float* bh2 = (const float*)d_in[8];
    const float* bi3 = (const float*)d_in[10];
    const float* bh3 = (const float*)d_in[12];
    const float* bo1 = (const float*)d_in[14];
    const float* bo2 = (const float*)d_in[16];
    const float* bo3 = (const float*)d_in[18];

    __bf16* wts    = (__bf16*)d_ws;
    float*  out    = (float*)d_out;
    float*  losses = out + (out_size - 3);

    // f32 -> bf16 weight conversion (idempotent, re-run every call)
    struct Seg { int src; unsigned off; int n; };
    const Seg segs[9] = {
        {1,  OFF_WI1, 512 * 64},  {3,  OFF_WH1, 512 * 512},
        {5,  OFF_WI2, 512 * 512}, {7,  OFF_WH2, 512 * 512},
        {9,  OFF_WI3, 512 * 512}, {11, OFF_WH3, 512 * 512},
        {13, OFF_WO1, 64 * 512},  {15, OFF_WO2, 64 * 512},
        {17, OFF_WO3, 64 * 512},
    };
    for (int s = 0; s < 9; ++s) {
        const int nseg = segs[s].n;
        cvt_f32_bf16<<<(nseg + 255) / 256, 256, 0, stream>>>(
            (const float*)d_in[segs[s].src], wts + segs[s].off, nseg);
    }
    zero_losses<<<1, 32, 0, stream>>>(losses);

    drnn_persistent<<<dim3(BATCH / 16), dim3(BLOCK), LDS_TOTAL, stream>>>(
        x, bi1, bh1, bi2, bh2, bi3, bh3, bo1, bo2, bo3, wts, out, losses);
}
